// DeepONet_3908420240249
// MI455X (gfx1250) — matrix operand order repaired and checked by reading, not compiled, not run on hardware
//
#include <hip/hip_runtime.h>
#include <hip/hip_bf16.h>
#include <math.h>

typedef __attribute__((ext_vector_type(16))) _Float16     v16h;
typedef __attribute__((ext_vector_type(8)))  _Float16     v8h;
typedef __attribute__((ext_vector_type(8)))  float        v8f;
typedef __attribute__((ext_vector_type(4)))  unsigned int v4u;
typedef __attribute__((ext_vector_type(8)))  unsigned int v8u;

#define SENSOR 128
#define BUNITS 1024
#define LORA   64
#define BATCH  2048
#define PTRUNK 33409   // total trunk params per sample

// ---------------------------------------------------------------------------
// Tensor Data Mover: 2D tile load Global -> LDS (CDNA5 ISA).
// Hand-packed D# groups 0/1, issued via inline asm (portable across both
// toolchains; avoids the 5-vs-6-arg builtin arity mismatch).
//   group0: [1:0]count=1 | [63:32]lds_addr | [120:64]global_addr | [127:126]type=2
//   group1: [17:16]data_size=1(2B) | tensor_dim0/1 | tile_dim0/1 | dim0_stride
// ---------------------------------------------------------------------------
__device__ __forceinline__ void tdm_load_tile_2d(
    const void* gsrc, unsigned lds_off,
    unsigned tdim0, unsigned tdim1,       // tensor extents (elements)
    unsigned tile0, unsigned tile1,       // tile extents (elements)
    unsigned stride0)                     // row stride (elements)
{
  unsigned long long ga = (unsigned long long)(uintptr_t)gsrc;
  v4u g0;
  g0[0] = 1u;                                            // count=1 (valid), user mode
  g0[1] = lds_off;                                       // LDS byte address
  g0[2] = (unsigned)ga;                                  // global_addr[31:0]
  g0[3] = (unsigned)((ga >> 32) & 0x01FFFFFFull)         // global_addr[56:32]
        | (2u << 30);                                    // type=2 ("image")
  v8u g1;
  g1[0] = 0x00010000u;                                   // data_size=1 -> 2 bytes/elem
  g1[1] = (tdim0 & 0xFFFFu) << 16;                       // tensor_dim0[15:0]
  g1[2] = (tdim0 >> 16) | ((tdim1 & 0xFFFFu) << 16);     // tensor_dim0[31:16], dim1[15:0]
  g1[3] = (tdim1 >> 16) | (tile0 << 16);                 // tensor_dim1[31:16], tile_dim0
  g1[4] = tile1 & 0xFFFFu;                               // tile_dim1 (tile_dim2=0)
  g1[5] = stride0;                                       // tensor_dim0_stride[31:0]
  g1[6] = 0u;
  g1[7] = 0u;
  asm volatile("tensor_load_to_lds %0, %1" :: "s"(g0), "s"(g1) : "memory");
}

// ---------------------------------------------------------------------------
// K0: fp32 -> f16 weight/activation conversion
// ---------------------------------------------------------------------------
__global__ void cvt_f32_f16(const float* __restrict__ s, _Float16* __restrict__ d, int n) {
  int i = blockIdx.x * blockDim.x + threadIdx.x;
  if (i < n) d[i] = (_Float16)s[i];
}

// K0b: trunk hidden-layer weight blocks in WMMA-B-friendly layout:
// W2t[l][r][o][i] = (f16) W2[r, Woff_l + i*128 + o],  l=0 -> off 384, l=1 -> off 16896
__global__ void build_w2t(const float* __restrict__ W2, _Float16* __restrict__ w2t) {
  int idx = blockIdx.x * blockDim.x + threadIdx.x;
  if (idx >= 2 * 64 * 128 * 128) return;
  int i = idx & 127, o = (idx >> 7) & 127, r = (idx >> 14) & 63, l = idx >> 20;
  int woff = l ? 16896 : 384;
  w2t[idx] = (_Float16)W2[(size_t)r * PTRUNK + woff + i * 128 + o];
}

// ---------------------------------------------------------------------------
// K1: tiled WMMA GEMM  C = act(A[MxK] @ B[KxN] + bias),  f16 in, f32 accum.
// Block tile 128x128, BK=32, 256 threads = 8 waves (4x2 wave grid, 32x64 each).
// A tile staged by the Tensor Data Mover (wave 0 issues, TENSORcnt wait),
// B tile transposed into LDS by all threads (overlaps the TDM copy).
// ---------------------------------------------------------------------------
template<bool TANH, bool OUTF32>
__global__ __launch_bounds__(256) void gemm_bias_act(
    const _Float16* __restrict__ A, const _Float16* __restrict__ B,
    const float* __restrict__ bias, void* __restrict__ Cout,
    int M, int N, int K)
{
  constexpr int BM = 128, BN = 128, BK = 32;
  __shared__ __align__(16) _Float16 sA[BM][BK];        // 8 KB, row-major (TDM dest)
  __shared__ __align__(16) _Float16 sBt[BN][BK + 8];   // transposed [n][k], padded
  const int tid  = threadIdx.x;
  const int lane = tid & 31, wave = tid >> 5;
  const int wm = wave >> 1, wn = wave & 1;             // wave grid 4(M) x 2(N)
  const int h = lane >> 4, l15 = lane & 15;
  const int m0 = blockIdx.x * BM, n0 = blockIdx.y * BN;
  const unsigned sA_base = (unsigned)(uintptr_t)&sA[0][0];

  v8f acc[2][4];
  #pragma unroll
  for (int i = 0; i < 2; ++i)
    #pragma unroll
    for (int j = 0; j < 4; ++j)
      #pragma unroll
      for (int e = 0; e < 8; ++e) acc[i][j][e] = 0.f;

  for (int k0 = 0; k0 < K; k0 += BK) {
    // ---- A tile (128x32 halves): async TDM copy, issued by wave 0 only
    if (wave == 0)
      tdm_load_tile_2d(A + (size_t)m0 * K + k0, sA_base,
                       (unsigned)K, (unsigned)BM, BK, BM, (unsigned)K);
    // ---- B tile transposed: sBt[n][k] = B[k0+k][n0+n], zero-pad n >= N
    {
      int kk = tid >> 3, nb = (tid & 7) * 16;
      const _Float16* src = B + (size_t)(k0 + kk) * N + n0 + nb;
      if (k0 + BK < K) __builtin_prefetch(src + (size_t)BK * N, 0, 1);
      #pragma unroll
      for (int j = 0; j < 16; ++j)
        sBt[nb + j][kk] = (n0 + nb + j < N) ? src[j] : (_Float16)0.f;
    }
    if (wave == 0) __builtin_amdgcn_s_wait_tensorcnt(0);
    __syncthreads();

    // ---- fragments (ISA 7.12.2 layouts)
    v16h aF[2], bF[4];
    #pragma unroll
    for (int i = 0; i < 2; ++i) {
      const _Float16* ap = &sA[wm * 32 + i * 16 + l15][0];
      v8h lo = *(const v8h*)&ap[8 * h];        // K = 8h..8h+7
      v8h hi = *(const v8h*)&ap[16 + 8 * h];   // K = 16+8h..23+8h
      #pragma unroll
      for (int e = 0; e < 8; ++e) { aF[i][e] = lo[e]; aF[i][8 + e] = hi[e]; }
    }
    #pragma unroll
    for (int j = 0; j < 4; ++j) {
      const _Float16* bp = &sBt[wn * 64 + j * 16 + l15][0];  // K = 8h..8h+7, 16+8h..23+8h
      v8h lo = *(const v8h*)(bp + 8 * h);
      v8h hi = *(const v8h*)(bp + 16 + 8 * h);
      #pragma unroll
      for (int e = 0; e < 8; ++e) { bF[j][e] = lo[e]; bF[j][8 + e] = hi[e]; }
    }
    #pragma unroll
    for (int i = 0; i < 2; ++i)
      #pragma unroll
      for (int j = 0; j < 4; ++j)
        acc[i][j] = __builtin_amdgcn_wmma_f32_16x16x32_f16(
            false, aF[i], false, bF[j], (short)0, acc[i][j], false, false);
    __syncthreads();
  }

  // ---- epilogue: bias (+tanh), store f16 or f32. C layout: M = v + 8h.
  #pragma unroll
  for (int i = 0; i < 2; ++i)
    #pragma unroll
    for (int j = 0; j < 4; ++j) {
      int col = n0 + wn * 64 + j * 16 + l15;
      if (col < N) {
        float bv = bias ? bias[col] : 0.f;
        #pragma unroll
        for (int v = 0; v < 8; ++v) {
          int row = m0 + wm * 32 + i * 16 + v + 8 * h;
          float val = acc[i][j][v] + bv;
          if (TANH) val = tanhf(val);
          if (OUTF32) ((float*)Cout)[(size_t)row * N + col] = val;
          else        ((_Float16*)Cout)[(size_t)row * N + col] = (_Float16)val;
        }
      }
    }
}

// ---------------------------------------------------------------------------
// K2: per-sample trunk. One workgroup = 16 samples, 8 waves.
// Hidden layer o-columns split across waves (wave w owns cols 16w..16w+15).
// For each r: Y[16,16] = X[16,128] @ W2blk[r][:,ocols]  (4 chained f16 WMMAs,
// B fragments loaded straight from L2-resident W2t), then
// acc[b,o] += z[b,r] * (Y[b,o] + biascol_r) in VALU on the fragments.
// ---------------------------------------------------------------------------
__global__ __launch_bounds__(256) void trunk_kernel(
    const float* __restrict__ t, const float* __restrict__ u,
    const float* __restrict__ Z, const float* __restrict__ W2,
    const _Float16* __restrict__ W2t, float* __restrict__ out)
{
  __shared__ float sZ[64][17];                       // sZ[r][m] = z[b0+m][r]
  __shared__ __align__(16) _Float16 sX[16][128];     // current activations (f16)
  __shared__ float red[16][16];
  const int tid  = threadIdx.x;
  const int lane = tid & 31, wave = tid >> 5;
  const int h = lane >> 4, l15 = lane & 15;
  const int b0 = blockIdx.x * 16;

  for (int i = tid; i < 16 * 64; i += 256) {
    int m = i & 15, r = i >> 4;
    sZ[r][m] = Z[(size_t)(b0 + m) * LORA + r];
  }
  __syncthreads();

  // ---- trunk layer 0:  x1[m,o] = tanh(t[m]*W0[m,o] + b0[m,o])
  for (int i = tid; i < 16 * 128; i += 256) {
    int o = i & 127, m = i >> 7;
    float wb = 0.f, ww = 0.f;
    #pragma unroll 8
    for (int r = 0; r < 64; ++r) {
      float zr = sZ[r][m];
      wb += zr * W2[(size_t)r * PTRUNK + o];          // bias col
      ww += zr * W2[(size_t)r * PTRUNK + 128 + o];    // weight col
    }
    sX[m][o] = (_Float16)tanhf(t[b0 + m] * ww + wb);
  }
  __syncthreads();

  // ---- trunk hidden layers 1 & 2 (128 -> 128)
  const int boffs[2] = {256, 16768};
  for (int l = 0; l < 2; ++l) {
    v16h aF[4];                          // X fragments, independent of r
    #pragma unroll
    for (int kk = 0; kk < 4; ++kk) {
      const _Float16* ap = &sX[l15][kk * 32];
      v8h lo = *(const v8h*)&ap[8 * h];
      v8h hi = *(const v8h*)&ap[16 + 8 * h];
      #pragma unroll
      for (int e = 0; e < 8; ++e) { aF[kk][e] = lo[e]; aF[kk][8 + e] = hi[e]; }
    }
    __syncthreads();   // sX fully consumed before it is overwritten below

    const int o = wave * 16 + l15;
    v8f acc;
    #pragma unroll
    for (int e = 0; e < 8; ++e) acc[e] = 0.f;

    const _Float16* wbase = W2t + ((size_t)l * 64) * 128 * 128 + (size_t)o * 128;
    for (int r = 0; r < 64; ++r) {
      const _Float16* bp = wbase + (size_t)r * 128 * 128;  // row o of W2t[l][r]
      v8f y;
      #pragma unroll
      for (int e = 0; e < 8; ++e) y[e] = 0.f;
      #pragma unroll
      for (int kk = 0; kk < 4; ++kk) {
        v16h bF;
        v8h lo = *(const v8h*)(bp + kk * 32 + 8 * h);      // K = 8h..8h+7
        v8h hi = *(const v8h*)(bp + kk * 32 + 16 + 8 * h);
        #pragma unroll
        for (int e = 0; e < 8; ++e) { bF[e] = lo[e]; bF[8 + e] = hi[e]; }
        y = __builtin_amdgcn_wmma_f32_16x16x32_f16(
            false, aF[kk], false, bF, (short)0, y, false, false);
      }
      float bval = W2[(size_t)r * PTRUNK + boffs[l] + o];  // fold bias into r-sum
      if (r + 1 < 64) __builtin_prefetch(bp + 128 * 128, 0, 1);
      #pragma unroll
      for (int v = 0; v < 8; ++v) {
        float zm = sZ[r][8 * h + v];                       // z[b = v+8h][r]
        acc[v] += zm * (y[v] + bval);
      }
    }
    #pragma unroll
    for (int v = 0; v < 8; ++v)
      sX[v + 8 * h][o] = (_Float16)tanhf(acc[v]);
    __syncthreads();
  }

  // ---- trunk layer 3 (128 -> 1) + DeepONet combine: out = q0 + t*q
  {
    int m = tid >> 4, s = tid & 15;     // 16 threads per sample, 8 i each
    float partial = 0.f;
    #pragma unroll
    for (int j = 0; j < 8; ++j) {
      int i = s * 8 + j;
      float w = 0.f;
      for (int r = 0; r < 64; ++r)
        w += sZ[r][m] * W2[(size_t)r * PTRUNK + 33281 + i];
      partial += (float)sX[m][i] * w;
    }
    red[m][s] = partial;
    __syncthreads();
    if (tid < 16) {
      int mm = tid;
      float q = 0.f;
      #pragma unroll
      for (int s2 = 0; s2 < 16; ++s2) q += red[mm][s2];
      float b3 = 0.f;
      for (int r = 0; r < 64; ++r)
        b3 += sZ[r][mm] * W2[(size_t)r * PTRUNK + 33280];
      q += b3;
      out[b0 + mm] = u[(size_t)(b0 + mm) * SENSOR] + t[b0 + mm] * q;
    }
  }
}

// ---------------------------------------------------------------------------
// host orchestration
// ---------------------------------------------------------------------------
extern "C" void kernel_launch(void* const* d_in, const int* in_sizes, int n_in,
                              void* d_out, int out_size, void* d_ws, size_t ws_size,
                              hipStream_t stream) {
  const float* t  = (const float*)d_in[0];
  const float* u  = (const float*)d_in[1];
  const float* bw[5] = {(const float*)d_in[2], (const float*)d_in[4],
                        (const float*)d_in[6], (const float*)d_in[8],
                        (const float*)d_in[10]};
  const float* bb[5] = {(const float*)d_in[3], (const float*)d_in[5],
                        (const float*)d_in[7], (const float*)d_in[9],
                        (const float*)d_in[11]};
  const float* W1 = (const float*)d_in[12];
  const float* W2 = (const float*)d_in[13];
  float* out = (float*)d_out;

  char* base = (char*)d_ws;
  size_t off = 0;
  auto halloc = [&](size_t elems) -> _Float16* {
    _Float16* p = (_Float16*)(base + off);
    off += ((elems * 2) + 255) & ~(size_t)255;
    return p;
  };
  _Float16* wU  = halloc((size_t)BATCH * SENSOR);
  _Float16* wB0 = halloc((size_t)SENSOR * BUNITS);
  _Float16* wBh[4];
  for (int i = 0; i < 4; ++i) wBh[i] = halloc((size_t)BUNITS * BUNITS);
  _Float16* wW1 = halloc((size_t)BUNITS * LORA);
  _Float16* W2t = halloc((size_t)2 * 64 * 128 * 128);
  _Float16* Ha  = halloc((size_t)BATCH * BUNITS);
  _Float16* Hb  = halloc((size_t)BATCH * BUNITS);
  float* Zf = (float*)(base + off);
  off += ((size_t)BATCH * LORA * 4 + 255) & ~(size_t)255;

  auto cvt = [&](const float* s, _Float16* d, int n) {
    cvt_f32_f16<<<(n + 255) / 256, 256, 0, stream>>>(s, d, n);
  };
  cvt(u, wU, BATCH * SENSOR);
  cvt(bw[0], wB0, SENSOR * BUNITS);
  for (int i = 0; i < 4; ++i) cvt(bw[i + 1], wBh[i], BUNITS * BUNITS);
  cvt(W1, wW1, BUNITS * LORA);
  build_w2t<<<(2 * 64 * 128 * 128 + 255) / 256, 256, 0, stream>>>(W2, W2t);

  dim3 blk(256);
  dim3 g1(BATCH / 128, BUNITS / 128);
  gemm_bias_act<true,  false><<<g1, blk, 0, stream>>>(wU, wB0,    bb[0], Ha, BATCH, BUNITS, SENSOR);
  gemm_bias_act<true,  false><<<g1, blk, 0, stream>>>(Ha, wBh[0], bb[1], Hb, BATCH, BUNITS, BUNITS);
  gemm_bias_act<true,  false><<<g1, blk, 0, stream>>>(Hb, wBh[1], bb[2], Ha, BATCH, BUNITS, BUNITS);
  gemm_bias_act<true,  false><<<g1, blk, 0, stream>>>(Ha, wBh[2], bb[3], Hb, BATCH, BUNITS, BUNITS);
  gemm_bias_act<false, false><<<g1, blk, 0, stream>>>(Hb, wBh[3], bb[4], Ha, BATCH, BUNITS, BUNITS);
  dim3 gz(BATCH / 128, 1);
  gemm_bias_act<false, true><<<gz, blk, 0, stream>>>(Ha, wW1, nullptr, Zf, BATCH, LORA, BUNITS);

  trunk_kernel<<<dim3(BATCH / 16), blk, 0, stream>>>(t, u, Zf, W2, W2t, out);
}
